// BlanchotianAttention_27221502722621
// MI455X (gfx1250) — matrix operand order repaired and checked by reading, not compiled, not run on hardware
//
#include <hip/hip_runtime.h>

typedef __attribute__((ext_vector_type(16))) __bf16 v16bf;
typedef __attribute__((ext_vector_type(8)))  __bf16 v8bf;
typedef __attribute__((ext_vector_type(4)))  __bf16 v4bf;
typedef __attribute__((ext_vector_type(8)))  float  v8f;

#define DIMN   1024
#define HEADS  16
#define HD     64
#define BATCH  2
#define SEQ    2048
#define ROWS   (BATCH*SEQ)   // 4096
#define N3     (3*DIMN)      // 3072
#define NP     2080          // padded key length (2049 -> 65*32)
#define NVALID 2049          // valid key count (seq + void)

static __device__ __forceinline__ v16bf cat16(v8bf lo, v8bf hi) {
  return __builtin_shufflevector(lo, hi, 0,1,2,3,4,5,6,7,8,9,10,11,12,13,14,15);
}
static __device__ __forceinline__ v8f wmma_bf16(v16bf a, v16bf b, v8f c) {
  return __builtin_amdgcn_wmma_f32_16x16x32_bf16(false, a, false, b, (short)0, c, false, false);
}

// ---- f32 -> bf16 elementwise convert (x) ------------------------------------
__global__ void k_cvt(const float* __restrict__ x, __bf16* __restrict__ xb, int n) {
  int i = (blockIdx.x * blockDim.x + threadIdx.x) * 4;
  if (i < n) {
    float4 f = *(const float4*)(x + i);
    v4bf o;
    o[0] = (__bf16)f.x; o[1] = (__bf16)f.y; o[2] = (__bf16)f.z; o[3] = (__bf16)f.w;
    *(v4bf*)(xb + i) = o;
  }
}

// ---- f32 [K][Ncols] -> bf16 transposed [Ncols][K] (K = 1024) ----------------
__global__ void k_transpose_w(const float* __restrict__ w, __bf16* __restrict__ wT, int ncols) {
  int idx = blockIdx.x * blockDim.x + threadIdx.x;   // over ncols*1024
  int k  = idx & 1023;
  int nn = idx >> 10;
  wT[idx] = (__bf16)w[(size_t)k * ncols + nn];
}

// ---- fill void token (row 2048) + zero pad rows 2049..2079 ------------------
__global__ void k_fill_pad(__bf16* __restrict__ qb, __bf16* __restrict__ kb,
                           __bf16* __restrict__ vT,
                           const float* __restrict__ vq, const float* __restrict__ vk,
                           const float* __restrict__ vv) {
  int idx = blockIdx.x * blockDim.x + threadIdx.x;   // 3 * 32 * 32 * 64 = 196608
  int which = idx >> 16;          // 0=q, 1=k, 2=vT
  int rem   = idx & 65535;
  int bh = rem >> 11;             // 0..31
  int rr = rem & 2047;
  int ro = rr >> 6;               // pad row offset 0..31
  int dd = rr & 63;
  int h  = bh & 15;
  int row = SEQ + ro;             // 2048..2079
  float val = 0.0f;
  if (row == SEQ) {
    const float* src = (which == 0) ? vq : (which == 1) ? vk : vv;
    val = src[h * HD + dd];
  }
  if (which == 2) {
    vT[((size_t)bh * HD + dd) * NP + row] = (__bf16)val;
  } else {
    __bf16* dst = which ? kb : qb;
    dst[((size_t)bh * NP + row) * HD + dd] = (__bf16)val;
  }
}

// ---- tiled bf16 WMMA GEMM: C[4096, N] = A[4096,1024] x Bm[N,1024]^T ---------
// MODE 0: scatter into q/k/vT buffers (N = 3072)
// MODE 1: f32 output + bias          (N = 1024)
template <int MODE>
__global__ void __launch_bounds__(256) k_gemm(
    const __bf16* __restrict__ A, const __bf16* __restrict__ Bm,
    __bf16* __restrict__ qb, __bf16* __restrict__ kb, __bf16* __restrict__ vT,
    float* __restrict__ outp, const float* __restrict__ bias)
{
  __shared__ __align__(16) __bf16 As[128 * 40];
  __shared__ __align__(16) __bf16 Bs[128 * 40];
  const int t    = threadIdx.x;
  const int lane = t & 31;
  // wave coordinates are uniform across the wave: force them scalar
  const int wave = __builtin_amdgcn_readfirstlane(t >> 5);
  const int wm   = wave & 3;     // 32-row group
  const int wn   = wave >> 2;    // 64-col group
  const int m0   = blockIdx.x * 128;
  const int n0   = blockIdx.y * 128;
  const int g16  = lane >> 4;
  const int l16  = lane & 15;
  const int lrow = t >> 1, lhalf = t & 1;

  v8f zero = {0.f,0.f,0.f,0.f,0.f,0.f,0.f,0.f};
  v8f acc[2][4];
#pragma unroll
  for (int i = 0; i < 2; i++)
#pragma unroll
    for (int j = 0; j < 4; j++) acc[i][j] = zero;

  for (int kk = 0; kk < DIMN; kk += 32) {
    { // stage 128x32 A and B tiles: each thread moves 32B of each
      const __bf16* gp = A  + (size_t)(m0 + lrow) * DIMN + kk + lhalf * 16;
      const __bf16* gq = Bm + (size_t)(n0 + lrow) * DIMN + kk + lhalf * 16;
      uint4 d0 = *(const uint4*)gp;
      uint4 d1 = *(const uint4*)(gp + 8);
      uint4 e0 = *(const uint4*)gq;
      uint4 e1 = *(const uint4*)(gq + 8);
      *(uint4*)&As[lrow * 40 + lhalf * 16]     = d0;
      *(uint4*)&As[lrow * 40 + lhalf * 16 + 8] = d1;
      *(uint4*)&Bs[lrow * 40 + lhalf * 16]     = e0;
      *(uint4*)&Bs[lrow * 40 + lhalf * 16 + 8] = e1;
    }
    __syncthreads();
    v16bf af[2], bfr[4];
#pragma unroll
    for (int mt = 0; mt < 2; mt++) {
      int row = wm * 32 + mt * 16 + l16;
      v8bf lo = *(const v8bf*)&As[row * 40 + g16 * 8];        // K = g16*8 + 0..7
      v8bf hi = *(const v8bf*)&As[row * 40 + 16 + g16 * 8];   // K = 16 + g16*8 + 0..7
      af[mt] = cat16(lo, hi);
    }
#pragma unroll
    for (int nt = 0; nt < 4; nt++) {
      int row = wn * 64 + nt * 16 + l16;
      v8bf lo = *(const v8bf*)&Bs[row * 40 + g16 * 8];        // K = g16*8 + 0..7
      v8bf hi = *(const v8bf*)&Bs[row * 40 + 16 + g16 * 8];   // K = 16 + g16*8 + 0..7
      bfr[nt] = cat16(lo, hi);
    }
#pragma unroll
    for (int mt = 0; mt < 2; mt++)
#pragma unroll
      for (int nt = 0; nt < 4; nt++)
        acc[mt][nt] = wmma_bf16(af[mt], bfr[nt], acc[mt][nt]);
    __syncthreads();
  }

  // ---- epilogue: all tile/head selection is wave-uniform (scalar) ----
  const int bidx = m0 >> 11;                    // batch index (tile never crosses)
#pragma unroll
  for (int nt = 0; nt < 4; nt++) {
    const int ncb = n0 + wn * 64 + nt * 16;     // scalar column base (16-aligned)
#pragma unroll
    for (int mt = 0; mt < 2; mt++) {
      const int mbase = (m0 & 2047) + wm * 32 + mt * 16;   // scalar seq base
      if (MODE == 1) {
        const int mg = m0 + wm * 32 + mt * 16;
#pragma unroll
        for (int r = 0; r < 8; r++) {
          int m = mg + g16 * 8 + r;
          int ncol = ncb + l16;
          outp[(size_t)m * DIMN + ncol] = acc[mt][nt][r] + bias[ncol];
        }
      } else {
        const int which = ncb >> 10;            // 0=q 1=k 2=v (scalar branch)
        const int h     = (ncb & 1023) >> 6;
        const int ddb   = ncb & 63;             // 16-aligned within head
        const int bh    = bidx * HEADS + h;
        if (which == 2) {
#pragma unroll
          for (int r = 0; r < 8; r++) {
            int nseq = mbase + g16 * 8 + r;
            vT[((size_t)bh * HD + ddb + l16) * NP + nseq] = (__bf16)acc[mt][nt][r];
          }
        } else {
          __bf16* dst = which ? kb : qb;
#pragma unroll
          for (int r = 0; r < 8; r++) {
            int nseq = mbase + g16 * 8 + r;
            dst[((size_t)bh * NP + nseq) * HD + ddb + l16] = (__bf16)acc[mt][nt][r];
          }
        }
      }
    }
  }
}

// ---- flash attention: 1 wave = 16 query rows x full head --------------------
__global__ void __launch_bounds__(128) k_attn(
    const __bf16* __restrict__ qb, const __bf16* __restrict__ kb,
    const __bf16* __restrict__ vT, __bf16* __restrict__ ab,
    const float* __restrict__ trace, const float* __restrict__ tfac)
{
  __shared__ __align__(16) __bf16 Pt[4][16 * 32];
  const int lane = threadIdx.x & 31;
  const int wid  = __builtin_amdgcn_readfirstlane(threadIdx.x >> 5);
  const int g    = blockIdx.x * 4 + wid;  // 0..4095 query tiles (scalar)
  const int bh   = g >> 7;
  const int qt   = g & 127;
  const int h    = bh & 15;
  const int g16  = lane >> 4, l16 = lane & 15;

  float temp   = fmaxf(1.0f + fabsf(trace[h]) * tfac[h], 1.0f);
  float sscale = 0.03125f / temp;          // dim^-0.5 / temperature

  // Q fragments for kdim 0..31 and 32..63 (held in registers for whole pass)
  const __bf16* qrow = qb + ((size_t)bh * NP + qt * 16 + l16) * HD;
  v16bf qf0 = cat16(*(const v8bf*)(qrow +      g16 * 16), *(const v8bf*)(qrow + 8 + g16 * 16));
  v16bf qf1 = cat16(*(const v8bf*)(qrow + 32 + g16 * 16), *(const v8bf*)(qrow + 40 + g16 * 16));

  float ms[8], ls[8];
#pragma unroll
  for (int r = 0; r < 8; r++) { ms[r] = -1e30f; ls[r] = 0.0f; }
  v8f zero = {0.f,0.f,0.f,0.f,0.f,0.f,0.f,0.f};
  v8f o[4];
#pragma unroll
  for (int d = 0; d < 4; d++) o[d] = zero;

  __bf16* pw = &Pt[wid][0];
  const __bf16* kbase = kb + (size_t)bh * NP * HD;
  const __bf16* vbase = vT + (size_t)bh * HD * NP;

  for (int jt = 0; jt < NP / 32; jt++) {
    const int j0 = jt * 32;
    // K fragments: key columns j0+l16 (S0) and j0+16+l16 (S1)
    const __bf16* kr0 = kbase + (size_t)(j0 + l16) * HD + g16 * 16;
    const __bf16* kr1 = kr0 + 16 * HD;
    v16bf bk0lo = cat16(*(const v8bf*)kr0,       *(const v8bf*)(kr0 + 8));
    v16bf bk0hi = cat16(*(const v8bf*)(kr0 + 32),*(const v8bf*)(kr0 + 40));
    v16bf bk1lo = cat16(*(const v8bf*)kr1,       *(const v8bf*)(kr1 + 8));
    v16bf bk1hi = cat16(*(const v8bf*)(kr1 + 32),*(const v8bf*)(kr1 + 40));

    v8f s0v = wmma_bf16(qf0, bk0lo, zero);
    s0v     = wmma_bf16(qf1, bk0hi, s0v);
    v8f s1v = wmma_bf16(qf0, bk1lo, zero);
    s1v     = wmma_bf16(qf1, bk1hi, s1v);

    // V fragments (vT row-contiguous in key index)
    v16bf vf[4];
#pragma unroll
    for (int d = 0; d < 4; d++) {
      const __bf16* vr = vbase + (size_t)(d * 16 + l16) * NP + j0 + g16 * 16;
      vf[d] = cat16(*(const v8bf*)vr, *(const v8bf*)(vr + 8));
    }

    const int col0 = j0 + l16;
    const int col1 = col0 + 16;
#pragma unroll
    for (int r = 0; r < 8; r++) {
      float s0 = s0v[r] * sscale;
      float s1 = s1v[r] * sscale;
      if (col0 >= NVALID) s0 = -1e30f;
      if (col1 >= NVALID) s1 = -1e30f;
      float rm = fmaxf(s0, s1);
      rm = fmaxf(rm, __shfl_xor(rm, 1));
      rm = fmaxf(rm, __shfl_xor(rm, 2));
      rm = fmaxf(rm, __shfl_xor(rm, 4));
      rm = fmaxf(rm, __shfl_xor(rm, 8));
      float mnew  = fmaxf(ms[r], rm);
      float alpha = __expf(ms[r] - mnew);
      float p0 = __expf(s0 - mnew);
      float p1 = __expf(s1 - mnew);
      float psum = p0 + p1;
      psum += __shfl_xor(psum, 1);
      psum += __shfl_xor(psum, 2);
      psum += __shfl_xor(psum, 4);
      psum += __shfl_xor(psum, 8);
      ls[r] = ls[r] * alpha + psum;
      ms[r] = mnew;
#pragma unroll
      for (int d = 0; d < 4; d++) o[d][r] *= alpha;
      int M = r + g16 * 8;                      // C-layout row
      pw[M * 32 + l16]      = (__bf16)p0;
      pw[M * 32 + 16 + l16] = (__bf16)p1;
    }
    // wave-internal LDS RAW: DS ops are in-order per wave, wait for the stores
    asm volatile("s_wait_dscnt 0" ::: "memory");
    // re-read P in A-fragment layout (M = l16, K = g16*16 + j)
    v8bf plo = *(const v8bf*)&pw[l16 * 32 + g16 * 16];
    v8bf phi = *(const v8bf*)&pw[l16 * 32 + g16 * 16 + 8];
    v16bf pf = cat16(plo, phi);
#pragma unroll
    for (int d = 0; d < 4; d++) o[d] = wmma_bf16(pf, vf[d], o[d]);
  }

  const int bidx = bh >> 4;
#pragma unroll
  for (int r = 0; r < 8; r++) {
    float inv = 1.0f / ls[r];
    int row = bidx * SEQ + qt * 16 + r + g16 * 8;
#pragma unroll
    for (int d = 0; d < 4; d++) {
      float v = o[d][r] * inv;
      ab[(size_t)row * DIMN + h * HD + d * 16 + l16] = (__bf16)v;
    }
  }
}

extern "C" void kernel_launch(void* const* d_in, const int* in_sizes, int n_in,
                              void* d_out, int out_size, void* d_ws, size_t ws_size,
                              hipStream_t stream) {
  const float* x      = (const float*)d_in[0];
  const float* w_qkv  = (const float*)d_in[1];
  const float* w_out  = (const float*)d_in[2];
  const float* b_out  = (const float*)d_in[3];
  const float* void_q = (const float*)d_in[4];
  const float* void_k = (const float*)d_in[5];
  const float* void_v = (const float*)d_in[6];
  const float* atrace = (const float*)d_in[7];
  const float* tfac   = (const float*)d_in[8];
  float* out = (float*)d_out;

  char* ws = (char*)d_ws;
  const size_t XB    = 0;                                   // 4096*1024 bf16 = 8 MB
  const size_t WQKVT = XB + (size_t)ROWS * DIMN * 2;        // 3072*1024 bf16 = 6 MB
  const size_t WOUTT = WQKVT + (size_t)N3 * DIMN * 2;       // 1024*1024 bf16 = 2 MB
  const size_t QKVB  = (size_t)BATCH * HEADS * NP * HD * 2; // 8,519,680 B each
  const size_t QB    = WOUTT + (size_t)DIMN * DIMN * 2;
  const size_t KB    = QB + QKVB;
  const size_t VT    = KB + QKVB;
  const size_t AB    = VT + QKVB;                           // 4096*1024 bf16 = 8 MB

  __bf16* xb    = (__bf16*)(ws + XB);
  __bf16* wqkvT = (__bf16*)(ws + WQKVT);
  __bf16* woutT = (__bf16*)(ws + WOUTT);
  __bf16* qb    = (__bf16*)(ws + QB);
  __bf16* kb    = (__bf16*)(ws + KB);
  __bf16* vT    = (__bf16*)(ws + VT);
  __bf16* ab    = (__bf16*)(ws + AB);

  // 1) precision conversion / weight transposition (one-time, L2-resident)
  k_cvt<<<(ROWS * DIMN) / 1024, 256, 0, stream>>>(x, xb, ROWS * DIMN);
  k_transpose_w<<<(N3 * DIMN) / 256, 256, 0, stream>>>(w_qkv, wqkvT, N3);
  k_transpose_w<<<(DIMN * DIMN) / 256, 256, 0, stream>>>(w_out, woutT, DIMN);
  k_fill_pad<<<(3 * 32 * 32 * HD) / 256, 256, 0, stream>>>(qb, kb, vT, void_q, void_k, void_v);

  // 2) QKV projection GEMM (scatter into head-major q/k/vT bf16 buffers)
  k_gemm<0><<<dim3(ROWS / 128, N3 / 128), 256, 0, stream>>>(
      xb, wqkvT, qb, kb, vT, nullptr, nullptr);

  // 3) flash attention (4096 query tiles, 4 waves per block)
  k_attn<<<(BATCH * HEADS * (SEQ / 16)) / 4, 128, 0, stream>>>(
      qb, kb, vT, ab, atrace, tfac);

  // 4) output projection GEMM + bias -> f32
  k_gemm<1><<<dim3(ROWS / 128, DIMN / 128), 256, 0, stream>>>(
      ab, woutT, nullptr, nullptr, nullptr, out, b_out);
}
